// MultiHeadAttentionBlock_1125281432317
// MI455X (gfx1250) — matrix operand order repaired and checked by reading, not compiled, not run on hardware
//
#include <hip/hip_runtime.h>
#include <hip/hip_bf16.h>
#include <math.h>

// ---------------------------------------------------------------------------
// MultiHeadAttentionBlock for MI455X (gfx1250, wave32, WMMA)
//   B=2, S=2048, D_MODEL=1024, H=16, D_K=64
// Pipeline (bf16 compute, f32 accumulate):
//   1) fp32 -> bf16 conversions
//   2) Q = q@Wq^T -> [B,H,S,64], K = k@Wk^T -> [B,H,S,64],
//      V = v@Wv^T -> [B,H,64,S] (transposed: direct WMMA-B operand)
//   3) flash attention: 4 waves/block share one (b,h); K/V tiles staged
//      into double-buffered LDS with GLOBAL_LOAD_ASYNC_TO_LDS_B128 +
//      s_wait_asynccnt (4x cut in L2 read traffic vs per-wave streams).
//   4) out = attn@Wo^T -> fp32 d_out
// ---------------------------------------------------------------------------

typedef __attribute__((ext_vector_type(4)))  __bf16 v4bf;
typedef __attribute__((ext_vector_type(8)))  __bf16 v8bf;
typedef __attribute__((ext_vector_type(16))) __bf16 v16bf;
typedef __attribute__((ext_vector_type(8)))  float  v8f;

#define S_LEN   2048
#define D_MODEL 1024
#define NHEAD   16
#define DK      64
#define MROWS   4096            // B*S
#define NEGVAL  (-1.0e9f)

// ---- WMMA wrapper: D = A(16x32 bf16) x B(32x16 bf16) + C(16x16 f32) --------
__device__ __forceinline__ v8f wmma_bf16(v16bf a, v16bf b, v8f c) {
  return __builtin_amdgcn_wmma_f32_16x16x32_bf16(false, a, false, b,
                                                 (short)0, c, false, false);
}

// ---- A-operand loader: row-major source, contiguous K.
// lane (m = lane&15, h = lane>>4) holds K = h*16+{0..15} (matches B loads).
__device__ __forceinline__ v16bf load_a_rowmajor(const __bf16* rowptr, int h) {
  v8bf lo = *(const v8bf*)(rowptr + h * 16);
  v8bf hi = *(const v8bf*)(rowptr + h * 16 + 8);
  return __builtin_shufflevector(lo, hi, 0, 1, 2, 3, 4, 5, 6, 7,
                                 8, 9, 10, 11, 12, 13, 14, 15);
}

// ---- async memory -> LDS (wave-wide, 16B/lane, tracked by ASYNCcnt) --------
// LDS operand is the raw LDS byte offset (low 32 bits of the generic
// address, aperture mapping per ISA 10.2), kept as a precomputed integer.
__device__ __forceinline__ void async_copy_b128(unsigned lds_off,
                                                const void* gptr) {
  asm volatile("global_load_async_to_lds_b128 %0, %1, off"
               :: "v"(lds_off), "v"(gptr) : "memory");
}

// ---- half-wave (16 lane) reductions: xor masks 1,2,4,8 stay inside halves
__device__ __forceinline__ float halfmax16(float v) {
  v = fmaxf(v, __shfl_xor(v, 1, 32));
  v = fmaxf(v, __shfl_xor(v, 2, 32));
  v = fmaxf(v, __shfl_xor(v, 4, 32));
  v = fmaxf(v, __shfl_xor(v, 8, 32));
  return v;
}
__device__ __forceinline__ float halfsum16(float v) {
  v += __shfl_xor(v, 1, 32);
  v += __shfl_xor(v, 2, 32);
  v += __shfl_xor(v, 4, 32);
  v += __shfl_xor(v, 8, 32);
  return v;
}

// ---------------------------------------------------------------------------
// fp32 -> bf16 conversion (vectorized x4; all sizes are multiples of 4)
// ---------------------------------------------------------------------------
__global__ void __launch_bounds__(256, 1)
cvt_f32_to_bf16(const float* __restrict__ src, __bf16* __restrict__ dst,
                int n4) {
  int i = blockIdx.x * blockDim.x + threadIdx.x;
  int stride = gridDim.x * blockDim.x;
  for (; i < n4; i += stride) {
    float4 v = ((const float4*)src)[i];
    v4bf o = {(__bf16)v.x, (__bf16)v.y, (__bf16)v.z, (__bf16)v.w};
    ((v4bf*)dst)[i] = o;
  }
}

// ---------------------------------------------------------------------------
// GEMM: C[m,n] = sum_k A[m,k] * W[n,k]   (x @ W^T)
//   A: [4096,1024] bf16 row-major, W: [1024,1024] bf16 row-major (L2 resident).
//   One wave computes a 16x64 output tile: 1 A-load + 4 B-loads + 4 WMMAs
//   per K-step of 32.  global_prefetch_b8 streams 128 elements ahead.
// MODE 0: store f32 [4096,1024]                  (output projection -> d_out)
// MODE 1: store bf16 head-split [B,H,S,64]       (Q, K)
// MODE 2: store bf16 head-split transposed [B,H,64,S]  (V)
// ---------------------------------------------------------------------------
template <int MODE>
__global__ void __launch_bounds__(256, 1)
gemm_xwT(const __bf16* __restrict__ A, const __bf16* __restrict__ W,
         void* __restrict__ out) {
  const int K = D_MODEL;
  const int gw = blockIdx.x * (blockDim.x >> 5) + (threadIdx.x >> 5);
  const int tileN = gw & 15;   // 1024/64 = 16 column tiles
  const int tileM = gw >> 4;   // 4096/16 = 256 row tiles
  const int lane = threadIdx.x & 31;
  const int ln = lane & 15;
  const int hh = lane >> 4;

  const __bf16* arow = A + (size_t)(tileM * 16 + ln) * K;
  const __bf16* wrow0 = W + (size_t)(tileN * 64 + 0 * 16 + ln) * K;
  const __bf16* wrow1 = W + (size_t)(tileN * 64 + 1 * 16 + ln) * K;
  const __bf16* wrow2 = W + (size_t)(tileN * 64 + 2 * 16 + ln) * K;
  const __bf16* wrow3 = W + (size_t)(tileN * 64 + 3 * 16 + ln) * K;

  v8f acc0 = {0.f, 0.f, 0.f, 0.f, 0.f, 0.f, 0.f, 0.f};
  v8f acc1 = acc0, acc2 = acc0, acc3 = acc0;

#pragma unroll 2
  for (int k0 = 0; k0 < K; k0 += 32) {
    if (k0 + 128 < K) {  // stream ahead -> global_prefetch_b8
      __builtin_prefetch(arow + k0 + 128, 0, 0);
      __builtin_prefetch(wrow0 + k0 + 128, 0, 0);
    }
    v16bf a = load_a_rowmajor(arow + k0, hh);
    v16bf b0 = *(const v16bf*)(wrow0 + k0 + hh * 16);
    v16bf b1 = *(const v16bf*)(wrow1 + k0 + hh * 16);
    v16bf b2 = *(const v16bf*)(wrow2 + k0 + hh * 16);
    v16bf b3 = *(const v16bf*)(wrow3 + k0 + hh * 16);
    acc0 = wmma_bf16(a, b0, acc0);
    acc1 = wmma_bf16(a, b1, acc1);
    acc2 = wmma_bf16(a, b2, acc2);
    acc3 = wmma_bf16(a, b3, acc3);
  }

  // C layout: VGPR r, lanes 0-15 -> (M=r, N=ln); lanes 16-31 -> (M=r+8, N=ln)
  v8f accs[4] = {acc0, acc1, acc2, acc3};
  if (MODE == 0) {
    float* o = (float*)out;
#pragma unroll
    for (int j = 0; j < 4; ++j) {
      int col = tileN * 64 + j * 16 + ln;
#pragma unroll
      for (int r = 0; r < 8; ++r) {
        int row = tileM * 16 + hh * 8 + r;
        o[(size_t)row * D_MODEL + col] = accs[j][r];
      }
    }
  } else {
    __bf16* o = (__bf16*)out;
#pragma unroll
    for (int j = 0; j < 4; ++j) {
      int col = tileN * 64 + j * 16 + ln;
      int head = col >> 6;
      int d = col & 63;
#pragma unroll
      for (int r = 0; r < 8; ++r) {
        int row = tileM * 16 + hh * 8 + r;
        int b = row >> 11;       // row / S_LEN
        int s = row & 2047;      // row % S_LEN
        size_t idx = (MODE == 1)
            ? ((size_t)((b * NHEAD + head) * S_LEN + s) * DK + d)
            : ((size_t)((b * NHEAD + head) * DK + d) * S_LEN + s);
        o[idx] = (__bf16)accs[j][r];
      }
    }
  }
}

// ---------------------------------------------------------------------------
// Flash attention.  Block = 4 waves sharing one (b,h), 4 adjacent 16-row
// query tiles.  Per 32-key block:
//   - K tile (32x64) and V tile (64x32, from transposed V) are async-DMAed
//     into double-buffered LDS (4 GLOBAL_LOAD_ASYNC_TO_LDS_B128 per wave),
//     synchronized with s_wait_asynccnt + workgroup barriers.
//   - 4 WMMAs (scores) + online softmax (half-wave ds_bpermute reductions)
//   - P relayout (C-layout -> A-layout) via per-wave LDS + s_wait_dscnt
//   - 4 WMMAs (P@V) accumulate the 16x64 output tile.
// All fill addresses are hoisted to loop-invariant bases + induction steps
// and the kernel is __launch_bounds__(128,1) to avoid VGPR spills.
// Q,K: [B,H,S,64] bf16;  Vt: [B,H,64,S] bf16;  out: [B,S,1024] bf16
// ---------------------------------------------------------------------------
__global__ void __launch_bounds__(128, 1)
flash_attn_kernel(const __bf16* __restrict__ Q, const __bf16* __restrict__ Kh,
                  const __bf16* __restrict__ Vt, const int* __restrict__ mask,
                  __bf16* __restrict__ outp) {
  __shared__ __align__(64) __bf16 sK[2][32][64];  // [buf][kseq][d]   4KB each
  __shared__ __align__(64) __bf16 sV[2][64][32];  // [buf][d][kseq]   4KB each
  __shared__ __align__(64) __bf16 sP[4][16][32];  // per-wave P staging

  const int w = threadIdx.x >> 5;
  const int tid = threadIdx.x;          // 0..127
  const int bh = blockIdx.x >> 5;       // b*16+h  (32 blocks per head)
  const int qt = (blockIdx.x & 31) * 4 + w;
  const int b = bh >> 4;
  const int h = bh & 15;
  const int lane = threadIdx.x & 31;
  const int ln = lane & 15;
  const int hh = lane >> 4;

  const __bf16* Qp = Q + ((size_t)bh * S_LEN + qt * 16) * DK;
  const __bf16* Kp = Kh + (size_t)bh * S_LEN * DK;
  const __bf16* Vp = Vt + (size_t)bh * DK * S_LEN;
  const int* mp = mask + b * S_LEN;

  // Q tile as two A operands (d = 0..31 and 32..63), loaded once
  const __bf16* qrow = Qp + ln * DK;
  v16bf qa0 = load_a_rowmajor(qrow, hh);
  v16bf qa1 = load_a_rowmajor(qrow + 32, hh);

  // ---- hoisted async-fill addressing: each thread owns two 16B chunks of
  // the K tile (32 rows x 8 chunks) and two of the V tile (64 rows x 4).
  // Per key block the K source advances 32*DK elements, V advances 32.
  const int c0 = tid, c1 = tid + 128;
  const __bf16* kg0 = Kp + (size_t)(c0 >> 3) * DK + (c0 & 7) * 8;
  const __bf16* kg1 = Kp + (size_t)(c1 >> 3) * DK + (c1 & 7) * 8;
  const __bf16* vg0 = Vp + (size_t)(c0 >> 2) * S_LEN + (c0 & 3) * 8;
  const __bf16* vg1 = Vp + (size_t)(c1 >> 2) * S_LEN + (c1 & 3) * 8;
  const unsigned kl0 = (unsigned)(unsigned long long)&sK[0][c0 >> 3][(c0 & 7) * 8];
  const unsigned kl1 = (unsigned)(unsigned long long)&sK[0][c1 >> 3][(c1 & 7) * 8];
  const unsigned vl0 = (unsigned)(unsigned long long)&sV[0][c0 >> 2][(c0 & 3) * 8];
  const unsigned vl1 = (unsigned)(unsigned long long)&sV[0][c1 >> 2][(c1 & 3) * 8];

  auto issue_fills = [&](int buf, int kbase) {
    const unsigned lofs = buf ? 4096u : 0u;  // double-buffer stride
    async_copy_b128(kl0 + lofs, kg0 + (size_t)kbase * DK);
    async_copy_b128(kl1 + lofs, kg1 + (size_t)kbase * DK);
    async_copy_b128(vl0 + lofs, vg0 + kbase);
    async_copy_b128(vl1 + lofs, vg1 + kbase);
  };

  float rmax[8], rsum[8];
  v8f zero = {0.f, 0.f, 0.f, 0.f, 0.f, 0.f, 0.f, 0.f};
  v8f acc[4] = {zero, zero, zero, zero};
#pragma unroll
  for (int r = 0; r < 8; ++r) { rmax[r] = -3.0e38f; rsum[r] = 0.f; }

  issue_fills(0, 0);  // prologue: stage first K/V tile

  for (int kb = 0; kb < S_LEN / 32; ++kb) {
    const int kbase = kb * 32;
    const int cur = kb & 1;

    // all waves done reading the buffer we are about to refill (iter kb-1)
    __syncthreads();
    if (kb + 1 < S_LEN / 32) {
      issue_fills(cur ^ 1, kbase + 32);
      // async loads complete in order: <=4 outstanding => current buffer done
      asm volatile("s_wait_asynccnt 0x4" ::: "memory");
    } else {
      asm volatile("s_wait_asynccnt 0x0" ::: "memory");
    }
    __syncthreads();  // publish current buffer from all waves

    // --- scores: S = Q K^T, K tile from LDS (B-operand: 32B per lane)
    v16bf kb0a = *(const v16bf*)&sK[cur][ln][hh * 16];
    v16bf kb0b = *(const v16bf*)&sK[cur][ln][32 + hh * 16];
    v16bf kb1a = *(const v16bf*)&sK[cur][16 + ln][hh * 16];
    v16bf kb1b = *(const v16bf*)&sK[cur][16 + ln][32 + hh * 16];
    v8f s0 = wmma_bf16(qa0, kb0a, zero);
    s0 = wmma_bf16(qa1, kb0b, s0);
    v8f s1 = wmma_bf16(qa0, kb1a, zero);
    s1 = wmma_bf16(qa1, kb1b, s1);

    const int mv0 = mp[kbase + ln];
    const int mv1 = mp[kbase + 16 + ln];

    // --- online softmax over this 32-key block
#pragma unroll
    for (int r = 0; r < 8; ++r) {
      float x0 = mv0 ? s0[r] * 0.125f : NEGVAL;  // 1/sqrt(64)
      float x1 = mv1 ? s1[r] * 0.125f : NEGVAL;
      float bm = halfmax16(fmaxf(x0, x1));
      float nm = fmaxf(rmax[r], bm);
      float corr = __expf(rmax[r] - nm);
      float p0 = __expf(x0 - nm);
      float p1 = __expf(x1 - nm);
      rsum[r] = rsum[r] * corr + halfsum16(p0 + p1);
      rmax[r] = nm;
      acc[0][r] *= corr;
      acc[1][r] *= corr;
      acc[2][r] *= corr;
      acc[3][r] *= corr;
      // C-layout -> LDS (row-major 16x32 bf16), cross-lane relayout
      sP[w][hh * 8 + r][ln] = (__bf16)p0;
      sP[w][hh * 8 + r][16 + ln] = (__bf16)p1;
    }

    // DS ops are wave-in-order; wait for stores before cross-lane reads
    asm volatile("s_wait_dscnt 0" ::: "memory");

    const __bf16* prow = &sP[w][ln][0];
    v8bf plo = *(const v8bf*)(prow + hh * 16);
    v8bf phi = *(const v8bf*)(prow + hh * 16 + 8);
    asm volatile("" ::: "memory");  // keep next-iter stores below these loads
    v16bf pa = __builtin_shufflevector(plo, phi, 0, 1, 2, 3, 4, 5, 6, 7,
                                       8, 9, 10, 11, 12, 13, 14, 15);

    // --- O += P @ V, V tile from LDS (B-operand: 32B per lane)
#pragma unroll
    for (int j = 0; j < 4; ++j) {
      v16bf vb = *(const v16bf*)&sV[cur][j * 16 + ln][hh * 16];
      acc[j] = wmma_bf16(pa, vb, acc[j]);
    }
  }

  // --- normalize and store to [B,S,H*64] bf16 for the output projection
  __bf16* obase = outp + (size_t)(b * S_LEN + qt * 16) * D_MODEL + h * DK;
#pragma unroll
  for (int r = 0; r < 8; ++r) {
    float inv = 1.0f / rsum[r];
    size_t rowoff = (size_t)(hh * 8 + r) * D_MODEL;
#pragma unroll
    for (int j = 0; j < 4; ++j) {
      obase[rowoff + j * 16 + ln] = (__bf16)(acc[j][r] * inv);
    }
  }
}

// ---------------------------------------------------------------------------
// Launch
// ---------------------------------------------------------------------------
extern "C" void kernel_launch(void* const* d_in, const int* in_sizes, int n_in,
                              void* d_out, int out_size, void* d_ws,
                              size_t ws_size, hipStream_t stream) {
  const float* q = (const float*)d_in[0];
  const float* k = (const float*)d_in[1];
  const float* v = (const float*)d_in[2];
  const int* mask = (const int*)d_in[3];
  const float* wq = (const float*)d_in[4];
  const float* wk = (const float*)d_in[5];
  const float* wv = (const float*)d_in[6];
  const float* wo = (const float*)d_in[7];

  const size_t SZ_X = (size_t)MROWS * D_MODEL * sizeof(__bf16);   // 8 MiB
  const size_t SZ_W = (size_t)D_MODEL * D_MODEL * sizeof(__bf16); // 2 MiB
  char* ws = (char*)d_ws;
  __bf16* qb  = (__bf16*)(ws);
  __bf16* kb  = (__bf16*)(ws + SZ_X);
  __bf16* vb  = (__bf16*)(ws + 2 * SZ_X);
  __bf16* wqb = (__bf16*)(ws + 3 * SZ_X);
  __bf16* wkb = (__bf16*)(ws + 3 * SZ_X + SZ_W);
  __bf16* wvb = (__bf16*)(ws + 3 * SZ_X + 2 * SZ_W);
  __bf16* wob = (__bf16*)(ws + 3 * SZ_X + 3 * SZ_W);
  __bf16* Qh  = (__bf16*)(ws + 3 * SZ_X + 4 * SZ_W);
  __bf16* Kh  = (__bf16*)(ws + 4 * SZ_X + 4 * SZ_W);
  __bf16* Vt  = (__bf16*)(ws + 5 * SZ_X + 4 * SZ_W);
  __bf16* Ab  = (__bf16*)(ws + 6 * SZ_X + 4 * SZ_W);  // attention output
  // total workspace use: 7*8MiB + 4*2MiB = 64 MiB

  // 1) fp32 -> bf16
  const int nx4 = MROWS * D_MODEL / 4;      // 1M vec4
  const int nw4 = D_MODEL * D_MODEL / 4;    // 256K vec4
  cvt_f32_to_bf16<<<(nx4 + 255) / 256, 256, 0, stream>>>(q, qb, nx4);
  cvt_f32_to_bf16<<<(nx4 + 255) / 256, 256, 0, stream>>>(k, kb, nx4);
  cvt_f32_to_bf16<<<(nx4 + 255) / 256, 256, 0, stream>>>(v, vb, nx4);
  cvt_f32_to_bf16<<<(nw4 + 255) / 256, 256, 0, stream>>>(wq, wqb, nw4);
  cvt_f32_to_bf16<<<(nw4 + 255) / 256, 256, 0, stream>>>(wk, wkb, nw4);
  cvt_f32_to_bf16<<<(nw4 + 255) / 256, 256, 0, stream>>>(wv, wvb, nw4);
  cvt_f32_to_bf16<<<(nw4 + 255) / 256, 256, 0, stream>>>(wo, wob, nw4);

  // 2) projections: 4096 wave-tiles, 8 waves/block -> 512 blocks x 256 thr
  gemm_xwT<1><<<512, 256, 0, stream>>>(qb, wqb, (void*)Qh);
  gemm_xwT<1><<<512, 256, 0, stream>>>(kb, wkb, (void*)Kh);
  gemm_xwT<2><<<512, 256, 0, stream>>>(vb, wvb, (void*)Vt);

  // 3) attention: 32 (b,h) pairs x 32 query groups -> 1024 blocks x 4 waves
  flash_attn_kernel<<<1024, 128, 0, stream>>>(Qh, Kh, Vt, mask, Ab);

  // 4) output projection -> fp32 d_out
  gemm_xwT<0><<<512, 256, 0, stream>>>(Ab, wob, d_out);
}
